// ModuleDistLayers_88794153877508
// MI455X (gfx1250) — matrix operand order repaired and checked by reading, not compiled, not run on hardware
//
#include <hip/hip_runtime.h>
#include <hip/hip_bf16.h>

// ---------------------------------------------------------------------------
// MI455X (gfx1250) implementation of the two dist-layers + final FC network.
// GEMMs run as bf16 WMMA (v_wmma_f32_16x16x32_bf16) with f32 accumulation:
// ~367 GFLOP of GEMM sits far below the ~90us HBM floor (~2 GB @ 23.3 TB/s),
// so the design goal is minimal streamed bytes (bf16 operands) and a fully
// overlapped global->LDS pipeline (async-to-LDS, ASYNCcnt) feeding WMMA.
// ---------------------------------------------------------------------------

typedef __attribute__((ext_vector_type(16))) __bf16 v16bf;
typedef __attribute__((ext_vector_type(8)))  __bf16 v8bf;
typedef __attribute__((ext_vector_type(8)))  float  v8f;
typedef int v4i __attribute__((vector_size(16)));   // matches builtin param type

#define BM   128
#define BN   128
#define BK   64      // two 16x16x32 K-substeps per LDS stage
#define BKP  72      // padded LDS row (144B): conflict-free b128 frag reads
#define NCOL 512
#define LDA  1280    // packed activation row stride

#if defined(__HIP_DEVICE_COMPILE__) && \
    __has_builtin(__builtin_amdgcn_global_load_async_to_lds_b128) && \
    __has_builtin(__builtin_amdgcn_s_wait_asynccnt)
#define ASYNC_COPY 1
// builtin signature (per hipcc diagnostic): (int4* src, int4* lds, Ii off, Ii cpol)
#define ASYNC_LD128(g, l) \
  __builtin_amdgcn_global_load_async_to_lds_b128((v4i*)(g), (v4i*)(l), 0, 0)
#endif

// ---------------------------------------------------------------------------
// Pack concat(x, dist_feat) -> bf16 A[N,1280]
__global__ __launch_bounds__(256)
void pack_x_kernel(const float* __restrict__ x, const float* __restrict__ dist,
                   __bf16* __restrict__ A, int n) {
  long idx = (long)blockIdx.x * blockDim.x + threadIdx.x;
  if (idx >= (long)n * LDA) return;
  int i = (int)(idx / LDA), c = (int)(idx % LDA);
  float v = (c < 1024) ? x[(long)i * 1024 + c] : dist[(long)i * 256 + (c - 1024)];
  A[idx] = (__bf16)v;
}

// Transpose-pack W[K,512] (fp32) -> Wt[512,LDA] (bf16, N-major, contiguous K)
__global__ __launch_bounds__(256)
void pack_wt_kernel(const float* __restrict__ W, __bf16* __restrict__ Wt, int K) {
  long idx = (long)blockIdx.x * blockDim.x + threadIdx.x;
  if (idx >= (long)NCOL * K) return;
  int nn = (int)(idx / K), k = (int)(idx % K);
  Wt[(long)nn * LDA + k] = (__bf16)W[(long)k * NCOL + nn];
}

// ---------------------------------------------------------------------------
// Hlin[N,512] = A_bf16[N,K] @ Wt_bf16[512,K]^T + bias ; fused per-column
// sum / sum-of-squares for BatchNorm batch statistics.
// Block tile 128x128, 8 wave32s, each wave 32x64 via 2x4 WMMA 16x16x32 frags,
// 16 WMMAs per barrier (BK=64), double-buffered LDS fed by async-to-LDS DMA.
__global__ __launch_bounds__(256)
void gemm_bias_stats_kernel(const __bf16* __restrict__ A,
                            const __bf16* __restrict__ Bt,
                            const float* __restrict__ bias,
                            float* __restrict__ Hlin,
                            float* __restrict__ colsum,
                            float* __restrict__ colsumsq,
                            int Nrows, int K) {
  __shared__ __bf16 As[2][BM][BKP];
  __shared__ __bf16 Bs[2][BM][BKP];
  __shared__ float  csum[BN];
  __shared__ float  csq[BN];

  const int t    = threadIdx.x;
  const int lane = t & 31;
  const int wave = t >> 5;
  const int wm   = wave & 3;          // 32-row strip within block tile
  const int wn   = wave >> 2;         // 64-col strip within block tile
  const int bm   = blockIdx.x * BM;
  const int bn   = blockIdx.y * BN;
  const int rsel = lane & 15;
  const int hi   = (lane >> 4) & 1;   // lane half selects K-group / M-half

  if (t < BN) { csum[t] = 0.f; csq[t] = 0.f; }

  // per-thread staging-chunk coordinates: 128 rows x 128B = 1024 x 16B chunks
  // per tile, 4 per thread.  (c>>3)=row, (c&7)*8 = element offset.
  int crow[4], coff[4];
#pragma unroll
  for (int i = 0; i < 4; ++i) {
    int c = t + i * 256;
    crow[i] = c >> 3;
    coff[i] = (c & 7) << 3;
  }
  const bool tail = (bm + BM > Nrows);

#ifdef ASYNC_COPY
  if (tail) {  // zero A staging so masked-off tail rows stay zero forever
    uint4 z = make_uint4(0u, 0u, 0u, 0u);
    for (int c = t; c < (int)(sizeof(As) / 16); c += 256) ((uint4*)As)[c] = z;
  }
#endif

  v8f acc[2][4];
#pragma unroll
  for (int mt = 0; mt < 2; ++mt)
#pragma unroll
    for (int nt = 0; nt < 4; ++nt) acc[mt][nt] = (v8f){0, 0, 0, 0, 0, 0, 0, 0};

  const int nk = K / BK;

  // ---- stage-0 load (global -> LDS)
#ifdef ASYNC_COPY
#pragma unroll
  for (int i = 0; i < 4; ++i) {
    int gr = bm + crow[i];
    if (gr < Nrows)
      ASYNC_LD128(A + (size_t)gr * LDA + coff[i], &As[0][crow[i]][coff[i]]);
    ASYNC_LD128(Bt + (size_t)(bn + crow[i]) * LDA + coff[i], &Bs[0][crow[i]][coff[i]]);
  }
  __builtin_amdgcn_s_wait_asynccnt(0);
#else
#pragma unroll
  for (int i = 0; i < 4; ++i) {
    uint4 va = make_uint4(0u, 0u, 0u, 0u);
    int gr = bm + crow[i];
    if (gr < Nrows) va = *(const uint4*)(A + (size_t)gr * LDA + coff[i]);
    *(uint4*)&As[0][crow[i]][coff[i]] = va;
    uint4 vb = *(const uint4*)(Bt + (size_t)(bn + crow[i]) * LDA + coff[i]);
    *(uint4*)&Bs[0][crow[i]][coff[i]] = vb;
  }
#endif
  __syncthreads();

  for (int kk = 0; kk < nk; ++kk) {
    const int cur = kk & 1;
    if (kk + 1 < nk) {                      // prefetch next K-slab
      const int nxt = cur ^ 1;
      const int k0  = (kk + 1) * BK;
#ifdef ASYNC_COPY
#pragma unroll
      for (int i = 0; i < 4; ++i) {
        int gr = bm + crow[i];
        if (gr < Nrows)
          ASYNC_LD128(A + (size_t)gr * LDA + k0 + coff[i], &As[nxt][crow[i]][coff[i]]);
        ASYNC_LD128(Bt + (size_t)(bn + crow[i]) * LDA + k0 + coff[i],
                    &Bs[nxt][crow[i]][coff[i]]);
      }
#else
#pragma unroll
      for (int i = 0; i < 4; ++i) {
        uint4 va = make_uint4(0u, 0u, 0u, 0u);
        int gr = bm + crow[i];
        if (gr < Nrows) va = *(const uint4*)(A + (size_t)gr * LDA + k0 + coff[i]);
        *(uint4*)&As[nxt][crow[i]][coff[i]] = va;
        uint4 vb = *(const uint4*)(Bt + (size_t)(bn + crow[i]) * LDA + k0 + coff[i]);
        *(uint4*)&Bs[nxt][crow[i]][coff[i]] = vb;
      }
#endif
    }

    // ---- two K=32 substeps from the current LDS stage
#pragma unroll
    for (int ks = 0; ks < 2; ++ks) {
      v16bf afr[2], bfr[4];
      const int ak = ks * 32 + (hi ? 8 : 0);   // A: ISA 7.12.2 interleave
#pragma unroll
      for (int mt = 0; mt < 2; ++mt) {
        int row = wm * 32 + mt * 16 + rsel;
        v8bf lo8 = *(const v8bf*)&As[cur][row][ak];
        v8bf hi8 = *(const v8bf*)&As[cur][row][ak + 16];
#pragma unroll
        for (int i = 0; i < 8; ++i) { afr[mt][i] = lo8[i]; afr[mt][i + 8] = hi8[i]; }
      }
      const int bk = ks * 32 + (hi ? 8 : 0);   // B: same interleave as A
#pragma unroll
      for (int nt = 0; nt < 4; ++nt) {
        int row = wn * 64 + nt * 16 + rsel;
        v8bf lo8 = *(const v8bf*)&Bs[cur][row][bk];
        v8bf hi8 = *(const v8bf*)&Bs[cur][row][bk + 16];
#pragma unroll
        for (int i = 0; i < 8; ++i) { bfr[nt][i] = lo8[i]; bfr[nt][i + 8] = hi8[i]; }
      }
#pragma unroll
      for (int mt = 0; mt < 2; ++mt)
#pragma unroll
        for (int nt = 0; nt < 4; ++nt)
          acc[mt][nt] = __builtin_amdgcn_wmma_f32_16x16x32_bf16(
              false, afr[mt], false, bfr[nt], (short)0, acc[mt][nt], false, false);
    }

#ifdef ASYNC_COPY
    __builtin_amdgcn_s_wait_asynccnt(0);
#endif
    __syncthreads();
  }

  // ---- epilogue: bias add, store, per-column BN partial stats
#pragma unroll
  for (int nt = 0; nt < 4; ++nt) {
    int lc   = wn * 64 + nt * 16 + rsel;
    int ncol = bn + lc;
    float b  = bias[ncol];
    float s = 0.f, s2 = 0.f;
#pragma unroll
    for (int mt = 0; mt < 2; ++mt) {
#pragma unroll
      for (int r = 0; r < 8; ++r) {
        int grow = bm + wm * 32 + mt * 16 + hi * 8 + r;
        if (grow < Nrows) {
          float v = acc[mt][nt][r] + b;
          Hlin[(size_t)grow * NCOL + ncol] = v;
          s += v; s2 += v * v;
        }
      }
    }
    atomicAdd(&csum[lc], s);
    atomicAdd(&csq[lc], s2);
  }
  __syncthreads();
  if (t < BN) {
    atomicAdd(&colsum[bn + t], csum[t]);
    atomicAdd(&colsumsq[bn + t], csq[t]);
  }
}

// ---------------------------------------------------------------------------
__global__ void bn_finalize_kernel(const float* __restrict__ colsum,
                                   const float* __restrict__ colsumsq,
                                   const float* __restrict__ g,
                                   const float* __restrict__ be,
                                   float* __restrict__ scale,
                                   float* __restrict__ shift, float invN) {
  int c = threadIdx.x;
  float m = colsum[c] * invN;
  float v = colsumsq[c] * invN - m * m;
  float s = g[c] * rsqrtf(v + 1e-5f);
  scale[c] = s;
  shift[c] = be[c] - m * s;
}

// BN + ReLU, write h (bf16) into A cols [0,512), accumulate segment sums
__global__ __launch_bounds__(256)
void bn_relu_seg_kernel(const float* __restrict__ Hlin,
                        const float* __restrict__ scale,
                        const float* __restrict__ shift,
                        const int* __restrict__ ele,
                        __bf16* __restrict__ A,
                        float* __restrict__ segsum,
                        float* __restrict__ segcnt, int n) {
  long idx = (long)blockIdx.x * blockDim.x + threadIdx.x;
  if (idx >= (long)n * NCOL) return;
  int i = (int)(idx >> 9), c = (int)(idx & 511);
  float v = scale[c] * Hlin[idx] + shift[c];
  v = v > 0.f ? v : 0.f;
  A[(size_t)i * LDA + c] = (__bf16)v;
  int e = ele[i];
  atomicAdd(&segsum[(size_t)e * NCOL + c], v);
  if (c == 0) atomicAdd(&segcnt[e], 1.0f);
}

// Gather segment means into A cols [512,1024)
__global__ __launch_bounds__(256)
void seg_gather_kernel(const float* __restrict__ segsum,
                       const float* __restrict__ segcnt,
                       const int* __restrict__ ele,
                       __bf16* __restrict__ A, int n) {
  long idx = (long)blockIdx.x * blockDim.x + threadIdx.x;
  if (idx >= (long)n * NCOL) return;
  int i = (int)(idx >> 9), c = (int)(idx & 511);
  int e = ele[i];
  float cnt = segcnt[e];
  cnt = cnt > 1.f ? cnt : 1.f;
  A[(size_t)i * LDA + 512 + c] = (__bf16)(segsum[(size_t)e * NCOL + c] / cnt);
}

// Final BN + ReLU -> fp32 output
__global__ __launch_bounds__(256)
void final_relu_kernel(const float* __restrict__ Hlin,
                       const float* __restrict__ scale,
                       const float* __restrict__ shift,
                       float* __restrict__ out, int n) {
  long idx = (long)blockIdx.x * blockDim.x + threadIdx.x;
  if (idx >= (long)n * NCOL) return;
  int c = (int)(idx & 511);
  float v = scale[c] * Hlin[idx] + shift[c];
  out[idx] = v > 0.f ? v : 0.f;
}

// ---------------------------------------------------------------------------
extern "C" void kernel_launch(void* const* d_in, const int* in_sizes, int n_in,
                              void* d_out, int out_size, void* d_ws, size_t ws_size,
                              hipStream_t stream) {
  const float* x    = (const float*)d_in[0];
  const float* dist = (const float*)d_in[1];
  const int*   ele  = (const int*)d_in[3];       // atom_idx (d_in[2]) unused
  const float* l1_w = (const float*)d_in[4];
  const float* l1_b = (const float*)d_in[5];
  const float* l1_g = (const float*)d_in[6];
  const float* l1_be= (const float*)d_in[7];
  const float* l2_w = (const float*)d_in[8];
  const float* l2_b = (const float*)d_in[9];
  const float* l2_g = (const float*)d_in[10];
  const float* l2_be= (const float*)d_in[11];
  const float* f_w  = (const float*)d_in[12];
  const float* f_b  = (const float*)d_in[13];
  const float* f_g  = (const float*)d_in[14];
  const float* f_be = (const float*)d_in[15];

  const int N = in_sizes[0] / 1024;              // rows (100000)

  // workspace carve-up
  char* ws = (char*)d_ws;
  size_t off = 0;
  __bf16* Abuf = (__bf16*)(ws + off); off += (size_t)N * LDA * 2;      // 256 MB
  float*  Hlin = (float*)(ws + off);  off += (size_t)N * NCOL * 4;     // 205 MB
  __bf16* Wt1  = (__bf16*)(ws + off); off += (size_t)NCOL * LDA * 2;
  __bf16* Wt2  = (__bf16*)(ws + off); off += (size_t)NCOL * LDA * 2;
  __bf16* Wt3  = (__bf16*)(ws + off); off += (size_t)NCOL * LDA * 2;
  float*  segsum = (float*)(ws + off); size_t segBytes = (size_t)1024 * NCOL * 4 + 4096;
  float*  segcnt = (float*)((char*)segsum + (size_t)1024 * NCOL * 4);
  off += segBytes;
  float* colsum   = (float*)(ws + off);          // colsum+colsumsq contiguous
  float* colsumsq = colsum + NCOL;
  off += 2 * NCOL * 4;
  float* scale = (float*)(ws + off); off += NCOL * 4;
  float* shift = (float*)(ws + off); off += NCOL * 4;

  const long nElems = (long)N * NCOL;
  const int  eBlks  = (int)((nElems + 255) / 256);
  dim3 gemmGrid((N + BM - 1) / BM, NCOL / BN);

  // ---- pack inputs / weights to bf16
  pack_x_kernel<<<(int)(((long)N * LDA + 255) / 256), 256, 0, stream>>>(x, dist, Abuf, N);
  pack_wt_kernel<<<(NCOL * 1280 + 255) / 256, 256, 0, stream>>>(l1_w, Wt1, 1280);
  pack_wt_kernel<<<(NCOL * 1280 + 255) / 256, 256, 0, stream>>>(l2_w, Wt2, 1280);
  pack_wt_kernel<<<(NCOL * 1024 + 255) / 256, 256, 0, stream>>>(f_w,  Wt3, 1024);

  // ---- layer 1
  (void)hipMemsetAsync(colsum, 0, 2 * NCOL * 4, stream);
  gemm_bias_stats_kernel<<<gemmGrid, 256, 0, stream>>>(Abuf, Wt1, l1_b, Hlin,
                                                       colsum, colsumsq, N, 1280);
  bn_finalize_kernel<<<1, NCOL, 0, stream>>>(colsum, colsumsq, l1_g, l1_be,
                                             scale, shift, 1.0f / (float)N);
  (void)hipMemsetAsync(segsum, 0, segBytes, stream);
  bn_relu_seg_kernel<<<eBlks, 256, 0, stream>>>(Hlin, scale, shift, ele, Abuf,
                                                segsum, segcnt, N);
  seg_gather_kernel<<<eBlks, 256, 0, stream>>>(segsum, segcnt, ele, Abuf, N);

  // ---- layer 2 (dist cols [1024,1280) in Abuf already hold dist_feat)
  (void)hipMemsetAsync(colsum, 0, 2 * NCOL * 4, stream);
  gemm_bias_stats_kernel<<<gemmGrid, 256, 0, stream>>>(Abuf, Wt2, l2_b, Hlin,
                                                       colsum, colsumsq, N, 1280);
  bn_finalize_kernel<<<1, NCOL, 0, stream>>>(colsum, colsumsq, l2_g, l2_be,
                                             scale, shift, 1.0f / (float)N);
  (void)hipMemsetAsync(segsum, 0, segBytes, stream);
  bn_relu_seg_kernel<<<eBlks, 256, 0, stream>>>(Hlin, scale, shift, ele, Abuf,
                                                segsum, segcnt, N);
  seg_gather_kernel<<<eBlks, 256, 0, stream>>>(segsum, segcnt, ele, Abuf, N);

  // ---- final FC (K=1024: h2 | mean2 gather) + BN + ReLU
  (void)hipMemsetAsync(colsum, 0, 2 * NCOL * 4, stream);
  gemm_bias_stats_kernel<<<gemmGrid, 256, 0, stream>>>(Abuf, Wt3, f_b, Hlin,
                                                       colsum, colsumsq, N, 1024);
  bn_finalize_kernel<<<1, NCOL, 0, stream>>>(colsum, colsumsq, f_g, f_be,
                                             scale, shift, 1.0f / (float)N);
  final_relu_kernel<<<eBlks, 256, 0, stream>>>(Hlin, scale, shift, (float*)d_out, N);
}
